// VarlenCrossAttentionWithMask_71708773974469
// MI455X (gfx1250) — matrix operand order repaired and checked by reading, not compiled, not run on hardware
//
#include <hip/hip_runtime.h>
#include <stdint.h>

// Problem constants (from reference): E=1024, H=16, dh=64, B=8, LQ=512, LKV=1024
#define E_DIM 1024
#define NHEAD 16
#define DHEAD 64
#define LQ_SEG 512
#define LKV_SEG 1024

typedef __attribute__((ext_vector_type(16))) __bf16 v16bf;
typedef __attribute__((ext_vector_type(8)))  __bf16 v8bf;
typedef __attribute__((ext_vector_type(4)))  __bf16 v4bf;
typedef __attribute__((ext_vector_type(8)))  float  v8f;
typedef unsigned short u16;

// ---- fragment helpers -------------------------------------------------------
// A-matrix 16x32 bf16 fragment (ISA 7.12.2): lane(l,half) holds row M=l,
// elements 0..7 = K k0+8*half+{0..7}, elements 8..15 = K k0+16+8*half+{0..7}.
// Caller passes p = rowbase + k0 + 8*half.
__device__ __forceinline__ v16bf frag_ldA(const u16* p) {
  v8bf lo = *reinterpret_cast<const v8bf*>(p);
  v8bf hi = *reinterpret_cast<const v8bf*>(p + 16);
  return __builtin_shufflevector(lo, hi, 0,1,2,3,4,5,6,7,8,9,10,11,12,13,14,15);
}
// B-matrix 32x16 bf16 fragment: lane(l,half) holds column N=l, elements 0..7 =
// K k0+8*half+{0..7}, 8..15 = K k0+16+8*half+{0..7}. Caller passes p = colbase + k0 + 8*half.
__device__ __forceinline__ v16bf frag_ldB(const u16* p) {
  v8bf lo = *reinterpret_cast<const v8bf*>(p);
  v8bf hi = *reinterpret_cast<const v8bf*>(p + 16);
  return __builtin_shufflevector(lo, hi, 0,1,2,3,4,5,6,7,8,9,10,11,12,13,14,15);
}
__device__ __forceinline__ v8f zero8() {
  v8f z;
#pragma unroll
  for (int i = 0; i < 8; ++i) z[i] = 0.0f;
  return z;
}

// ---- fp32 -> bf16 conversion ------------------------------------------------
__global__ void cvt_f32_bf16(const float* __restrict__ s, u16* __restrict__ d, int n) {
  int i = (blockIdx.x * blockDim.x + threadIdx.x) * 4;
  if (i < n) {
    float4 v = *reinterpret_cast<const float4*>(s + i);
    v4bf o;
    o[0] = (__bf16)v.x; o[1] = (__bf16)v.y; o[2] = (__bf16)v.z; o[3] = (__bf16)v.w;
    *reinterpret_cast<v4bf*>(d + i) = o;
  }
}

// ---- GEMM: Y = X(M x 1024) * W(1024 x 1024)^T + bias ------------------------
// Wave tile 32x64 (2x4 wmma accumulators). 8 waves / 256-thread block.
// MODE 0: Y bf16 row-major [M x 1024]
// MODE 1: Y bf16 per-head transposed Vt[(seg*H+h)*64+d][LKV]  (for P*V B-frags)
// MODE 2: Y fp32 row-major (final output)
// MODE is a template parameter so each instantiation has a branch-free
// epilogue (the runtime-branch version spilled VGPRs across the K loop).
template <int MODE>
__global__ void gemm_bf16_wmma(const u16* __restrict__ X, const u16* __restrict__ W,
                               const float* __restrict__ bias,
                               u16* __restrict__ Ybf, float* __restrict__ Yf,
                               int M) {
  const int lane = threadIdx.x & 31;
  const int half = lane >> 4;
  const int l    = lane & 15;
  const int wid  = threadIdx.x >> 5;
  const int tilesM = M >> 5;
  int gid = blockIdx.x * 8 + wid;
  int tm = gid % tilesM, tn = gid / tilesM;
  const int m0 = tm * 32, n0 = tn * 64;

  v8f acc[2][4];
#pragma unroll
  for (int i = 0; i < 2; ++i)
#pragma unroll
    for (int j = 0; j < 4; ++j) acc[i][j] = zero8();

  for (int k0 = 0; k0 < E_DIM; k0 += 32) {
    v16bf a[2], bm[4];
#pragma unroll
    for (int i = 0; i < 2; ++i)
      a[i] = frag_ldA(X + (size_t)(m0 + i * 16 + l) * E_DIM + k0 + half * 8);
#pragma unroll
    for (int j = 0; j < 4; ++j)
      bm[j] = frag_ldB(W + (size_t)(n0 + j * 16 + l) * E_DIM + k0 + half * 8);
#pragma unroll
    for (int i = 0; i < 2; ++i)
#pragma unroll
      for (int j = 0; j < 4; ++j)
        acc[i][j] = __builtin_amdgcn_wmma_f32_16x16x32_bf16(
            false, a[i], false, bm[j], (short)0, acc[i][j], false, false);
  }

  __bf16* yb = reinterpret_cast<__bf16*>(Ybf);
#pragma unroll
  for (int i = 0; i < 2; ++i) {
#pragma unroll
    for (int j = 0; j < 4; ++j) {
      const int ncol = n0 + j * 16 + l;
      const float bv = bias[ncol];
#pragma unroll
      for (int r = 0; r < 8; ++r) {
        const int m = m0 + i * 16 + half * 8 + r;
        const float val = acc[i][j][r] + bv;
        if (MODE == 0) {
          yb[(size_t)m * E_DIM + ncol] = (__bf16)val;
        } else if (MODE == 1) {
          const int seg = m >> 10;        // / LKV_SEG
          const int tok = m & (LKV_SEG - 1);
          const int h = ncol >> 6, d = ncol & 63;
          yb[((size_t)((seg * NHEAD + h) * DHEAD + d)) * LKV_SEG + tok] = (__bf16)val;
        } else {
          Yf[(size_t)m * E_DIM + ncol] = val;
        }
      }
    }
  }
}

// ---- Flash attention: one wave per (b, h, 16-query tile) --------------------
// S^T = K_chunk * Q^T  -> D layout (lane=q column, regs=kv rows) is exactly
// the A-fragment layout of P for O += P * V. Online softmax in fp32.
__global__ void attn_kernel(const u16* __restrict__ Qp, const u16* __restrict__ Kp,
                            const u16* __restrict__ Vt, u16* __restrict__ Aout) {
  const int lane = threadIdx.x & 31;
  const int half = lane >> 4;
  const int l    = lane & 15;
  const int wid  = threadIdx.x >> 5;
  const int g  = blockIdx.x * 8 + wid;   // 0..4095
  const int qt = g & 31;                 // LQ/16 tiles
  const int h  = (g >> 5) & 15;
  const int b  = g >> 9;

  // Q^T B-fragments (dh x 16), reused for all kv chunks
  const u16* qbase = Qp + (size_t)(b * LQ_SEG + qt * 16 + l) * E_DIM + h * DHEAD;
  v16bf qf[2];
#pragma unroll
  for (int j = 0; j < 2; ++j) qf[j] = frag_ldB(qbase + j * 32 + half * 8);

  float mrun = -__builtin_inff();
  float lrun = 0.0f;
  v8f oacc[4];
#pragma unroll
  for (int t = 0; t < 4; ++t) oacc[t] = zero8();

  const float scale = 0.125f; // 1/sqrt(64)
  const u16* khead = Kp + (size_t)(b * LKV_SEG) * E_DIM + h * DHEAD;
  const u16* vhead = Vt + (size_t)((b * NHEAD + h) * DHEAD) * LKV_SEG;

  for (int c = 0; c < LKV_SEG; c += 32) {
    // Prefetch next chunk's K rows / V columns (wraps at the end; harmless).
    {
      const int cn = (c + 32) & (LKV_SEG - 1);
      __builtin_prefetch(khead + (size_t)(cn + l) * E_DIM, 0, 0);
      __builtin_prefetch(khead + (size_t)(cn + 16 + l) * E_DIM, 0, 0);
      __builtin_prefetch(vhead + (size_t)l * LKV_SEG + cn, 0, 0);
    }
    // S^T chunk: two 16(kv) x 16(q) subtiles, K-dim = dh = 64 (2 wmma each)
    v8f st[2];
#pragma unroll
    for (int s = 0; s < 2; ++s) {
      v8f a = zero8();
      const u16* kbase = khead + (size_t)(c + s * 16 + l) * E_DIM;
#pragma unroll
      for (int j = 0; j < 2; ++j) {
        v16bf kf = frag_ldA(kbase + j * 32 + half * 8);
        a = __builtin_amdgcn_wmma_f32_16x16x32_bf16(false, kf, false, qf[j],
                                                    (short)0, a, false, false);
      }
      st[s] = a;
    }
    // scale + chunk max per q column (pair lanes l and l^16 hold same q)
    float mx = -__builtin_inff();
#pragma unroll
    for (int s = 0; s < 2; ++s)
#pragma unroll
      for (int r = 0; r < 8; ++r) { st[s][r] *= scale; mx = fmaxf(mx, st[s][r]); }
    mx = fmaxf(mx, __shfl_xor(mx, 16));
    const float mnew = fmaxf(mrun, mx);
    const float corr = __expf(mrun - mnew);
    mrun = mnew;
    float rsum = 0.0f;
#pragma unroll
    for (int s = 0; s < 2; ++s)
#pragma unroll
      for (int r = 0; r < 8; ++r) {
        const float e = __expf(st[s][r] - mnew);
        st[s][r] = e;
        rsum += e;
      }
    rsum += __shfl_xor(rsum, 16);
    lrun = lrun * corr + rsum;
    // rescale O rows (row q = 8*half + r; corr for q lives in lane q)
#pragma unroll
    for (int r = 0; r < 8; ++r) {
      const float cr = __shfl(corr, (half << 3) + r);
#pragma unroll
      for (int t = 0; t < 4; ++t) oacc[t][r] *= cr;
    }
    // P A-fragment: st[0] -> kv {8h+0..7}, st[1] -> kv {16+8h+0..7} : exact match
    v16bf pf;
#pragma unroll
    for (int e = 0; e < 8; ++e) { pf[e] = (__bf16)st[0][e]; pf[e + 8] = (__bf16)st[1][e]; }
    // O += P(16q x 32kv) * V(32kv x 64dh), V^T rows contiguous in kv
#pragma unroll
    for (int t = 0; t < 4; ++t) {
      v16bf vf = frag_ldB(vhead + (size_t)(t * 16 + l) * LKV_SEG + c + half * 8);
      oacc[t] = __builtin_amdgcn_wmma_f32_16x16x32_bf16(false, pf, false, vf,
                                                        (short)0, oacc[t], false, false);
    }
  }
  // final 1/l normalization
  const float inv = 1.0f / lrun;
#pragma unroll
  for (int r = 0; r < 8; ++r) {
    const float ir = __shfl(inv, (half << 3) + r);
#pragma unroll
    for (int t = 0; t < 4; ++t) oacc[t][r] *= ir;
  }
  __bf16* out = reinterpret_cast<__bf16*>(Aout);
#pragma unroll
  for (int t = 0; t < 4; ++t)
#pragma unroll
    for (int r = 0; r < 8; ++r) {
      const int token = b * LQ_SEG + qt * 16 + half * 8 + r;
      out[(size_t)token * E_DIM + h * DHEAD + t * 16 + l] = (__bf16)oacc[t][r];
    }
}

// ---- launch -----------------------------------------------------------------
extern "C" void kernel_launch(void* const* d_in, const int* in_sizes, int n_in,
                              void* d_out, int out_size, void* d_ws, size_t ws_size,
                              hipStream_t stream) {
  const float* inputs  = (const float*)d_in[0];  // [8192,1024]
  const float* queries = (const float*)d_in[1];  // [4096,1024]
  const float* w_in    = (const float*)d_in[2];  // [3072,1024]
  const float* b_in    = (const float*)d_in[3];  // [3072]
  const float* w_out   = (const float*)d_in[4];  // [1024,1024]
  const float* b_out   = (const float*)d_in[5];  // [1024]
  float* out = (float*)d_out;                    // [4096,1024] fp32

  // Workspace carve-up (bf16 elements); total ~84 MB.
  u16* ws = (u16*)d_ws;
  size_t off = 0;
  u16* qb  = ws + off; off += (size_t)4096 * 1024;  // queries bf16
  u16* xb  = ws + off; off += (size_t)8192 * 1024;  // inputs bf16
  u16* wb  = ws + off; off += (size_t)3072 * 1024;  // w_in bf16 (q;k;v rows)
  u16* wob = ws + off; off += (size_t)1024 * 1024;  // w_out bf16
  u16* Qp  = ws + off; off += (size_t)4096 * 1024;  // projected Q bf16
  u16* Kp  = ws + off; off += (size_t)8192 * 1024;  // projected K bf16
  u16* Vt  = ws + off; off += (size_t)8192 * 1024;  // projected V^T per head
  u16* Ao  = ws + off; off += (size_t)4096 * 1024;  // attention output bf16

  auto cvt = [&](const float* s, u16* d, int n) {
    cvt_f32_bf16<<<(n / 4 + 255) / 256, 256, 0, stream>>>(s, d, n);
  };
  cvt(queries, qb, 4096 * 1024);
  cvt(inputs,  xb, 8192 * 1024);
  cvt(w_in,    wb, 3072 * 1024);
  cvt(w_out,   wob, 1024 * 1024);

  // QKV projections (tiles = (M/32)*(1024/64); 8 wave-tiles per block)
  gemm_bf16_wmma<0><<<256, 256, 0, stream>>>(qb, wb,                   b_in,        Qp, nullptr, 4096);
  gemm_bf16_wmma<0><<<512, 256, 0, stream>>>(xb, wb + 1024 * 1024,     b_in + 1024, Kp, nullptr, 8192);
  gemm_bf16_wmma<1><<<512, 256, 0, stream>>>(xb, wb + 2 * 1024 * 1024, b_in + 2048, Vt, nullptr, 8192);

  // Attention: 8*16*(512/16) = 4096 wave tiles = 512 blocks x 8 waves
  attn_kernel<<<512, 256, 0, stream>>>(Qp, Kp, Vt, Ao);

  // Output projection -> fp32 d_out
  gemm_bf16_wmma<2><<<256, 256, 0, stream>>>(Ao, wob, b_out, nullptr, out, 4096);
}
